// Transformer_9242769621769
// MI455X (gfx1250) — compile-verified, matrix operand order checked by reading, not run on hardware
//
#include <hip/hip_runtime.h>

#define NPTS 65536
#define NSAMP 16
#define CIN 64
#define COUT 64
#define MIDS 8
#define RTOT (NPTS*NSAMP)
#define EPSV 1e-5f

typedef __attribute__((ext_vector_type(16))) _Float16 v16h;
typedef __attribute__((ext_vector_type(8)))  _Float16 v8h;
typedef __attribute__((ext_vector_type(8)))  float    v8f;

__device__ __forceinline__ float relu_f(float x) { return x > 0.f ? x : 0.f; }

// Build a 16x32 f16 A/B fragment from a row-major fp32 matrix with ld=64 (global).
// ISA layout (16-bit A 16x32): lane group g holds K = {8g..8g+7} in v16h[0..7]
// and K = {16+8g..16+8g+7} in v16h[8..15] for each 32-wide K chunk.
__device__ __forceinline__ v16h load_frag64(const float* __restrict__ rowbase,
                                            int kchunk, int g) {
  const float4* p4 = reinterpret_cast<const float4*>(rowbase + kchunk * 32);
  float4 a0 = p4[2 * g + 0];
  float4 a1 = p4[2 * g + 1];
  float4 a2 = p4[2 * g + 4];
  float4 a3 = p4[2 * g + 5];
  v16h f;
  f[0] = (_Float16)a0.x;  f[1] = (_Float16)a0.y;  f[2]  = (_Float16)a0.z;  f[3]  = (_Float16)a0.w;
  f[4] = (_Float16)a1.x;  f[5] = (_Float16)a1.y;  f[6]  = (_Float16)a1.z;  f[7]  = (_Float16)a1.w;
  f[8] = (_Float16)a2.x;  f[9] = (_Float16)a2.y;  f[10] = (_Float16)a2.z;  f[11] = (_Float16)a2.w;
  f[12] = (_Float16)a3.x; f[13] = (_Float16)a3.y; f[14] = (_Float16)a3.z;  f[15] = (_Float16)a3.w;
  return f;
}

// Same fragment, but from an f16 row staged in LDS: two 16B ds_load_b128 per frag.
__device__ __forceinline__ v16h lds_frag64(const _Float16* rowp, int kchunk, int g) {
  const v8h* pv = reinterpret_cast<const v8h*>(rowp + kchunk * 32 + 8 * g);
  v8h lo = pv[0];   // K = kchunk*32 + 8g .. +7
  v8h hi = pv[2];   // K = kchunk*32 + 8g + 16 .. +7
  v16h f;
#pragma unroll
  for (int j = 0; j < 8; ++j) { f[j] = lo[j]; f[8 + j] = hi[j]; }
  return f;
}

// Cooperative f32 -> f16 staging of a 64x64 weight matrix into LDS (once per block).
__device__ __forceinline__ void stage_w64(const float* __restrict__ W, _Float16* dst) {
  for (int i = threadIdx.x; i < (COUT * CIN) / 4; i += 256) {
    float4 v = reinterpret_cast<const float4*>(W)[i];
    _Float16* d = dst + i * 4;
    d[0] = (_Float16)v.x; d[1] = (_Float16)v.y; d[2] = (_Float16)v.z; d[3] = (_Float16)v.w;
  }
}

// ---------------- Phase 1: q/k/v projections via WMMA (weights staged in LDS) ----------------
__device__ __forceinline__ void proj_one(const _Float16* wl,
                                         const float* __restrict__ bias,
                                         float* __restrict__ out,
                                         v16h a0, v16h a1,
                                         int rbase, int g, int ln) {
#pragma unroll
  for (int ct = 0; ct < 4; ++ct) {
    int n = ct * 16 + ln;                       // output column for this lane
    v16h b0 = lds_frag64(wl + n * CIN, 0, g);   // B[k][n] = W[n][k]
    v16h b1 = lds_frag64(wl + n * CIN, 1, g);
    v8f acc = {};
    acc = __builtin_amdgcn_wmma_f32_16x16x32_f16(false, a0, false, b0, (short)0, acc, false, false);
    acc = __builtin_amdgcn_wmma_f32_16x16x32_f16(false, a1, false, b1, (short)0, acc, false, false);
    float bv = bias[n];
#pragma unroll
    for (int j = 0; j < 8; ++j)                 // D: vgpr j -> row 8g+j, col = ln
      out[(size_t)(rbase + 8 * g + j) * COUT + n] = acc[j] + bv;
  }
}

__global__ void __launch_bounds__(256) qkv_gemm_kernel(
    const float* __restrict__ x,
    const float* __restrict__ Wq, const float* __restrict__ bq,
    const float* __restrict__ Wk, const float* __restrict__ bk,
    const float* __restrict__ Wv, const float* __restrict__ bv,
    float* __restrict__ xq, float* __restrict__ xk, float* __restrict__ xv) {
  __shared__ _Float16 wlds[3 * COUT * CIN];     // 24 KB: f16 weights, shared by 8 waves
  stage_w64(Wq, wlds);
  stage_w64(Wk, wlds + COUT * CIN);
  stage_w64(Wv, wlds + 2 * COUT * CIN);
  __syncthreads();

  int wave = threadIdx.x >> 5;
  int lane = threadIdx.x & 31;
  int g = lane >> 4, ln = lane & 15;
  int rbase = (blockIdx.x * 8 + wave) * 16;
  int m = rbase + ln;
  v16h a0 = load_frag64(x + (size_t)m * CIN, 0, g);
  v16h a1 = load_frag64(x + (size_t)m * CIN, 1, g);
  proj_one(wlds,                 bq, xq, a0, a1, rbase, g, ln);
  proj_one(wlds + COUT * CIN,    bk, xk, a0, a1, rbase, g, ln);
  proj_one(wlds + 2 * COUT * CIN,bv, xv, a0, a1, rbase, g, ln);
}

// ---------------- Phase 2: p_r1 = (p[idx]-p) @ Wp1^T + bp1, + 3-ch stats ----------------
__global__ void __launch_bounds__(256) prel_kernel(
    const float* __restrict__ p, const int* __restrict__ idx,
    const float* __restrict__ Wp1, const float* __restrict__ bp1,
    float* __restrict__ p_r1, float* __restrict__ psum, float* __restrict__ psq) {
  __shared__ float sh[6];
  if (threadIdx.x < 6) sh[threadIdx.x] = 0.f;
  __syncthreads();
  int r = blockIdx.x * 256 + threadIdx.x;   // grid is exact: r < RTOT
  int n = r >> 4;
  int j = idx[r];
  float d0 = p[j * 3 + 0] - p[n * 3 + 0];
  float d1 = p[j * 3 + 1] - p[n * 3 + 1];
  float d2 = p[j * 3 + 2] - p[n * 3 + 2];
#pragma unroll
  for (int i = 0; i < 3; ++i) {
    float o = Wp1[i * 3 + 0] * d0 + Wp1[i * 3 + 1] * d1 + Wp1[i * 3 + 2] * d2 + bp1[i];
    p_r1[(size_t)r * 3 + i] = o;
    atomicAdd(&sh[i], o);
    atomicAdd(&sh[3 + i], o * o);
  }
  __syncthreads();
  if (threadIdx.x < 6) {
    float* dst = (threadIdx.x < 3) ? (psum + threadIdx.x) : (psq + threadIdx.x - 3);
    atomicAdd(dst, sh[threadIdx.x]);
  }
}

// ---------------- BN finalize: scale = g*rstd, shift = beta - mean*scale ----------------
__global__ void finalize_kernel(const float* __restrict__ sum, const float* __restrict__ sq,
                                const float* __restrict__ g, const float* __restrict__ beta,
                                float* __restrict__ scale, float* __restrict__ shift,
                                int C, float invM) {
  int c = threadIdx.x;
  if (c < C) {
    float m = sum[c] * invM;
    float v = sq[c] * invM - m * m;
    float rs = rsqrtf(v + EPSV);
    float sc = g[c] * rs;
    scale[c] = sc;
    shift[c] = beta[c] - m * sc;
  }
}

// ---------------- Phase 3: global stats of w_pre (recomputed, never stored) ----------------
__global__ void __launch_bounds__(256) wstats_kernel(
    const float* __restrict__ xq, const float* __restrict__ xk,
    const float* __restrict__ p_r1, const int* __restrict__ idx,
    const float* __restrict__ Wp2, const float* __restrict__ bp2,
    const float* __restrict__ pscale, const float* __restrict__ pshift,
    float* __restrict__ wsum, float* __restrict__ wsq) {
  __shared__ float ssum[64], ssq[64];
  if (threadIdx.x < 64) { ssum[threadIdx.x] = 0.f; ssq[threadIdx.x] = 0.f; }
  __syncthreads();
  int t = blockIdx.x * blockDim.x + threadIdx.x;
  int kb = (t & 7) * 8;                         // this thread's 8-channel slab
  int row0 = t >> 3;
  int rstride = (gridDim.x * blockDim.x) >> 3;
  float s[8] = {0, 0, 0, 0, 0, 0, 0, 0};
  float q[8] = {0, 0, 0, 0, 0, 0, 0, 0};
  float ps0 = pscale[0], ps1 = pscale[1], ps2 = pscale[2];
  float pb0 = pshift[0], pb1 = pshift[1], pb2 = pshift[2];
  for (int r = row0; r < RTOT; r += rstride) {
    int n = r >> 4;
    int jrow = idx[r];
    const float* kr = xk + (size_t)jrow * COUT;   // L2-resident gathers
    const float* qr = xq + (size_t)n * COUT;
    const float* pr = p_r1 + (size_t)r * 3;
    float pa0 = relu_f(pr[0] * ps0 + pb0);
    float pa1 = relu_f(pr[1] * ps1 + pb1);
    float pa2 = relu_f(pr[2] * ps2 + pb2);
#pragma unroll
    for (int j = 0; j < 8; ++j) {
      int k = kb + j;
      float pr2 = pa0 * Wp2[k * 3 + 0] + pa1 * Wp2[k * 3 + 1] + pa2 * Wp2[k * 3 + 2] + bp2[k];
      float wv = kr[k] - qr[k] + pr2;
      s[j] += wv;
      q[j] += wv * wv;
    }
  }
#pragma unroll
  for (int j = 0; j < 8; ++j) {
    atomicAdd(&ssum[kb + j], s[j]);
    atomicAdd(&ssq[kb + j], q[j]);
  }
  __syncthreads();
  if (threadIdx.x < 64) {
    atomicAdd(&wsum[threadIdx.x], ssum[threadIdx.x]);
    atomicAdd(&wsq[threadIdx.x], ssq[threadIdx.x]);
  }
}

// ---------------- Phase 4: h1 = relu(bn(w_pre)) @ Ww1^T + bw1 via WMMA, + 8-ch stats ----------------
__global__ void __launch_bounds__(256) h1_kernel(
    const float* __restrict__ xq, const float* __restrict__ xk,
    const float* __restrict__ p_r1, const int* __restrict__ idx,
    const float* __restrict__ Wp2, const float* __restrict__ bp2,
    const float* __restrict__ Ww1, const float* __restrict__ bw1,
    const float* __restrict__ pscale, const float* __restrict__ pshift,
    const float* __restrict__ wscale, const float* __restrict__ wshift,
    float* __restrict__ h1, float* __restrict__ hstats /* [16]: sum8 | sq8 */) {
  __shared__ _Float16 w1lds[16 * CIN];    // Ww1^T zero-padded to 16 output cols (f16)
  __shared__ float sWp2[CIN * 3];         // per-k constants, reused 64x per lane
  __shared__ float sbp2[CIN], sws[CIN], swsh[CIN];
  __shared__ float sh[16];

  if (threadIdx.x < 16) sh[threadIdx.x] = 0.f;
  for (int i = threadIdx.x; i < 16 * CIN; i += 256) {
    int c = i >> 6, k = i & 63;
    w1lds[i] = (_Float16)((c < MIDS) ? Ww1[c * CIN + k] : 0.f);
  }
  if (threadIdx.x < CIN * 3) sWp2[threadIdx.x] = Wp2[threadIdx.x];
  if (threadIdx.x < CIN) {
    sbp2[threadIdx.x] = bp2[threadIdx.x];
    sws[threadIdx.x]  = wscale[threadIdx.x];
    swsh[threadIdx.x] = wshift[threadIdx.x];
  }
  __syncthreads();

  int wave = threadIdx.x >> 5;
  int lane = threadIdx.x & 31;
  int g = lane >> 4, ln = lane & 15;
  int rbase = (blockIdx.x * 8 + wave) * 16;
  int r = rbase + ln;                         // this lane's A-matrix row (an (n,ns) pair)
  int n = r >> 4;
  int jrow = idx[r];
  const float* kr = xk + (size_t)jrow * COUT;
  const float* qr = xq + (size_t)n * COUT;
  const float* pr = p_r1 + (size_t)r * 3;
  float pa0 = relu_f(pr[0] * pscale[0] + pshift[0]);
  float pa1 = relu_f(pr[1] * pscale[1] + pshift[1]);
  float pa2 = relu_f(pr[2] * pscale[2] + pshift[2]);

  // A fragments: relu(bn(w_pre)) computed on the fly in f16
  v16h afrag[2];
#pragma unroll
  for (int kc = 0; kc < 2; ++kc) {
    v16h f;
#pragma unroll
    for (int half = 0; half < 2; ++half) {
      int kb = kc * 32 + 8 * g + 16 * half;
#pragma unroll
      for (int j = 0; j < 8; ++j) {
        int k = kb + j;
        float pr2 = pa0 * sWp2[k * 3 + 0] + pa1 * sWp2[k * 3 + 1] + pa2 * sWp2[k * 3 + 2] + sbp2[k];
        float wv = kr[k] - qr[k] + pr2;
        f[half * 8 + j] = (_Float16)relu_f(wv * sws[k] + swsh[k]);
      }
    }
    afrag[kc] = f;
  }

  // B fragments straight from LDS (padding already baked in)
  int c = ln;
  v16h b0 = lds_frag64(w1lds + c * CIN, 0, g);
  v16h b1 = lds_frag64(w1lds + c * CIN, 1, g);

  v8f acc = {};
  acc = __builtin_amdgcn_wmma_f32_16x16x32_f16(false, afrag[0], false, b0, (short)0, acc, false, false);
  acc = __builtin_amdgcn_wmma_f32_16x16x32_f16(false, afrag[1], false, b1, (short)0, acc, false, false);

  if (c < MIDS) {
    float bias = bw1[c];
    float s = 0.f, sq = 0.f;
#pragma unroll
    for (int j = 0; j < 8; ++j) {
      float v = acc[j] + bias;
      h1[(size_t)(rbase + 8 * g + j) * MIDS + c] = v;
      s += v;
      sq += v * v;
    }
    atomicAdd(&sh[c], s);
    atomicAdd(&sh[8 + c], sq);
  }
  __syncthreads();
  if (threadIdx.x < 16) atomicAdd(&hstats[threadIdx.x], sh[threadIdx.x]);
}

// ---------------- Phase 5: bn/relu -> Ww2 -> softmax(ns) -> weighted aggregation ----------------
__global__ void __launch_bounds__(256) out_kernel(
    const float* __restrict__ xv, const float* __restrict__ p_r1,
    const float* __restrict__ h1, const int* __restrict__ idx,
    const float* __restrict__ Wp2, const float* __restrict__ bp2,
    const float* __restrict__ pscale, const float* __restrict__ pshift,
    const float* __restrict__ hscale, const float* __restrict__ hshift,
    const float* __restrict__ Ww2, const float* __restrict__ bw2,
    float* __restrict__ out) {
  int gid = blockIdx.x * 256 + threadIdx.x;   // one thread per (point, softmax-channel c<8)
  int n = gid >> 3;
  int c = gid & 7;
  float w2r[8], hs[8], hb[8];
#pragma unroll
  for (int j = 0; j < 8; ++j) {
    w2r[j] = Ww2[c * 8 + j];
    hs[j] = hscale[j];
    hb[j] = hshift[j];
  }
  float bias = bw2[c];
  float att[16];
  float mx = -1e30f;
#pragma unroll
  for (int ns = 0; ns < 16; ++ns) {
    const float* hr = h1 + (size_t)(n * 16 + ns) * MIDS;
    float acc = bias;
#pragma unroll
    for (int j = 0; j < 8; ++j) acc += relu_f(hr[j] * hs[j] + hb[j]) * w2r[j];
    att[ns] = acc;
    mx = fmaxf(mx, acc);
  }
  float ssum = 0.f;
#pragma unroll
  for (int ns = 0; ns < 16; ++ns) {
    float e = __expf(att[ns] - mx);
    att[ns] = e;
    ssum += e;
  }
  float inv = 1.f / ssum;
  float acc8[8] = {0, 0, 0, 0, 0, 0, 0, 0};
  float ps0 = pscale[0], ps1 = pscale[1], ps2 = pscale[2];
  float pb0 = pshift[0], pb1 = pshift[1], pb2 = pshift[2];
  for (int ns = 0; ns < 16; ++ns) {
    int r = n * 16 + ns;
    int jrow = idx[r];
    const float* pr = p_r1 + (size_t)r * 3;
    float pa0 = relu_f(pr[0] * ps0 + pb0);
    float pa1 = relu_f(pr[1] * ps1 + pb1);
    float pa2 = relu_f(pr[2] * ps2 + pb2);
    float w = att[ns] * inv;
    const float* xvr = xv + (size_t)jrow * COUT;
#pragma unroll
    for (int s = 0; s < 8; ++s) {
      int k = s * 8 + c;
      float pr2 = pa0 * Wp2[k * 3 + 0] + pa1 * Wp2[k * 3 + 1] + pa2 * Wp2[k * 3 + 2] + bp2[k];
      acc8[s] += (xvr[k] + pr2) * w;
    }
  }
#pragma unroll
  for (int s = 0; s < 8; ++s) out[(size_t)n * COUT + s * 8 + c] = acc8[s];
}

__global__ void zero_kernel(float* __restrict__ ptr, int n) {
  int i = blockIdx.x * blockDim.x + threadIdx.x;
  if (i < n) ptr[i] = 0.f;
}

extern "C" void kernel_launch(void* const* d_in, const int* in_sizes, int n_in,
                              void* d_out, int out_size, void* d_ws, size_t ws_size,
                              hipStream_t stream) {
  (void)in_sizes; (void)n_in; (void)out_size; (void)ws_size;
  const float* p      = (const float*)d_in[0];
  const float* x      = (const float*)d_in[1];
  const int*   idx    = (const int*)d_in[2];
  const float* Wq     = (const float*)d_in[3];
  const float* bq     = (const float*)d_in[4];
  const float* Wk     = (const float*)d_in[5];
  const float* bk     = (const float*)d_in[6];
  const float* Wv     = (const float*)d_in[7];
  const float* bv     = (const float*)d_in[8];
  const float* Wp1    = (const float*)d_in[9];
  const float* bp1    = (const float*)d_in[10];
  const float* gp     = (const float*)d_in[11];
  const float* betap  = (const float*)d_in[12];
  const float* Wp2    = (const float*)d_in[13];
  const float* bp2    = (const float*)d_in[14];
  const float* gw1    = (const float*)d_in[15];
  const float* betaw1 = (const float*)d_in[16];
  const float* Ww1    = (const float*)d_in[17];
  const float* bw1    = (const float*)d_in[18];
  const float* gw2    = (const float*)d_in[19];
  const float* betaw2 = (const float*)d_in[20];
  const float* Ww2    = (const float*)d_in[21];
  const float* bw2    = (const float*)d_in[22];
  float* out = (float*)d_out;

  // Workspace layout (floats): xq|xk|xv (16.8MB each) | p_r1 (12.6MB) | h1 (33.5MB) | stats
  float* ws    = (float*)d_ws;
  float* xq    = ws;
  float* xk    = xq + (size_t)NPTS * COUT;
  float* xv    = xk + (size_t)NPTS * COUT;
  float* p_r1  = xv + (size_t)NPTS * COUT;
  float* h1    = p_r1 + (size_t)RTOT * 3;
  float* stats = h1 + (size_t)RTOT * MIDS;
  float* psum   = stats;         // 3
  float* psq    = stats + 3;     // 3
  float* wsum   = stats + 6;     // 64
  float* wsq    = stats + 70;    // 64
  float* hstats = stats + 134;   // 16 (sum8 | sq8)
  float* fin    = stats + 150;
  float* pscale = fin;           // 3
  float* pshift = fin + 3;       // 3
  float* wscale = fin + 6;       // 64
  float* wshift = fin + 70;      // 64
  float* hscale = fin + 134;     // 8
  float* hshift = fin + 142;     // 8  (fin total: 150)

  const float invM = 1.0f / (float)RTOT;

  zero_kernel<<<2, 256, 0, stream>>>(stats, 300);
  qkv_gemm_kernel<<<NPTS / 128, 256, 0, stream>>>(x, Wq, bq, Wk, bk, Wv, bv, xq, xk, xv);
  prel_kernel<<<RTOT / 256, 256, 0, stream>>>(p, idx, Wp1, bp1, p_r1, psum, psq);
  finalize_kernel<<<1, 64, 0, stream>>>(psum, psq, gp, betap, pscale, pshift, 3, invM);
  wstats_kernel<<<2048, 256, 0, stream>>>(xq, xk, p_r1, idx, Wp2, bp2, pscale, pshift, wsum, wsq);
  finalize_kernel<<<1, 64, 0, stream>>>(wsum, wsq, gw1, betaw1, wscale, wshift, 64, invM);
  h1_kernel<<<RTOT / 128, 256, 0, stream>>>(xq, xk, p_r1, idx, Wp2, bp2, Ww1, bw1,
                                            pscale, pshift, wscale, wshift, h1, hstats);
  finalize_kernel<<<1, 64, 0, stream>>>(hstats, hstats + 8, gw2, betaw2, hscale, hshift, MIDS, invM);
  out_kernel<<<(NPTS * 8) / 256, 256, 0, stream>>>(xv, p_r1, h1, idx, Wp2, bp2,
                                                   pscale, pshift, hscale, hshift, Ww2, bw2, out);
}
